// CausalSelfAttention_8254927143087
// MI455X (gfx1250) — matrix operand order repaired and checked by reading, not compiled, not run on hardware
//
#include <hip/hip_runtime.h>
#include <hip/hip_bf16.h>

// CDNA5 / gfx1250, wave32. bf16 WMMA with f32 accumulate + async global->LDS.

typedef __attribute__((ext_vector_type(16))) __bf16 v16bf;
typedef __attribute__((ext_vector_type(8)))  __bf16 v8bf;
typedef __attribute__((ext_vector_type(2)))  __bf16 v2bf;
typedef __attribute__((ext_vector_type(8)))  float  v8f;
typedef __attribute__((ext_vector_type(4)))  float  v4f;

#define WMMA_BF16(A_, B_, C_) \
  __builtin_amdgcn_wmma_f32_16x16x32_bf16(false, (A_), false, (B_), (short)0, (C_), false, false)

// Problem constants: B=2, T=2048, C=1024, H=16, D=64
#define TT   2048
#define CC_  1024
#define HH   16
#define DD   64

static __device__ __forceinline__ void st_pk_f(__bf16* p, float a, float b) {
  v2bf t; t[0] = (__bf16)a; t[1] = (__bf16)b;
  *(v2bf*)p = t;                       // one ds_store_b32
}
static __device__ __forceinline__ void st_pk_bf(__bf16* p, __bf16 a, __bf16 b) {
  v2bf t; t[0] = a; t[1] = b;
  *(v2bf*)p = t;
}

// ---------------------------------------------------------------------------
// Kernel 1: qkv = x @ w_attn + b_attn -> Q (scaled 1/8), K, V in [B,H,T,D] bf16
// Block tile 128x128, 8 waves, wave tile 32x64 (2x4 WMMA tiles), K-step 32.
// ---------------------------------------------------------------------------
__global__ __launch_bounds__(256) void qkv_gemm_kernel(
    const float* __restrict__ x, const float* __restrict__ w,
    const float* __restrict__ bias,
    __bf16* __restrict__ Qo, __bf16* __restrict__ Ko, __bf16* __restrict__ Vo)
{
  constexpr int LDA = 40, LDB = 40;
  __shared__ __align__(16) __bf16 As[128 * LDA];      // [m][k]
  __shared__ __align__(16) __bf16 Bs[128 * LDB];      // transposed [n][k]

  const int tid  = threadIdx.x;
  const int lane = tid & 31;
  const int wave = tid >> 5;
  const int wy   = wave & 3;        // 4 row groups of 32
  const int wx   = wave >> 2;       // 2 col groups of 64
  const int l16  = lane & 15;
  const int half = lane >> 4;
  const int m0 = blockIdx.y * 128;
  const int n0 = blockIdx.x * 128;

  v8f acc[2][4] = {};

  for (int kk = 0; kk < 1024; kk += 32) {
    // A: 128x32 f32 -> bf16 LDS, pack pairs along k (b32 stores).
#pragma unroll
    for (int p = 0; p < 4; ++p) {
      int row = p * 32 + (tid >> 3);
      int col = (tid & 7) * 4;
      v4f xa = *(const v4f*)(x + (size_t)(m0 + row) * 1024 + (kk + col));
      st_pk_f(&As[row * LDA + col],     xa[0], xa[1]);
      st_pk_f(&As[row * LDA + col + 2], xa[2], xa[3]);
    }
    // B: 32x128 f32, transposed [n][k]; load two k-rows, pack (k,k+1) pairs.
#pragma unroll
    for (int p = 0; p < 2; ++p) {
      int c   = p * 256 + tid;        // 512 tasks: 16 k-pairs x 32 col-groups
      int kp  = c >> 5;               // k-pair 0..15
      int col = (c & 31) * 4;
      const float* wp = w + (size_t)(kk + 2 * kp) * 3072 + (n0 + col);
      v4f r0 = *(const v4f*)wp;
      v4f r1 = *(const v4f*)(wp + 3072);
#pragma unroll
      for (int j = 0; j < 4; ++j)
        st_pk_f(&Bs[(col + j) * LDB + 2 * kp], r0[j], r1[j]);
    }
    __syncthreads();

    v16bf afr[2], bfr[4];
#pragma unroll
    for (int i = 0; i < 2; ++i) {
      int m   = wy * 32 + i * 16 + l16;
      int klo = half * 8;
      ((v8bf*)&afr[i])[0] = *(const v8bf*)&As[m * LDA + klo];
      ((v8bf*)&afr[i])[1] = *(const v8bf*)&As[m * LDA + klo + 16];
    }
#pragma unroll
    for (int j = 0; j < 4; ++j) {
      int n  = wx * 64 + j * 16 + l16;
      int kb = half * 8;
      ((v8bf*)&bfr[j])[0] = *(const v8bf*)&Bs[n * LDB + kb];
      ((v8bf*)&bfr[j])[1] = *(const v8bf*)&Bs[n * LDB + kb + 16];
    }
#pragma unroll
    for (int i = 0; i < 2; ++i)
#pragma unroll
      for (int j = 0; j < 4; ++j)
        acc[i][j] = WMMA_BF16(afr[i], bfr[j], acc[i][j]);

    __syncthreads();
  }

#pragma unroll
  for (int i = 0; i < 2; ++i)
#pragma unroll
    for (int j = 0; j < 4; ++j) {
      int n   = n0 + wx * 64 + j * 16 + l16;
      int sec = n >> 10;              // 0=Q 1=K 2=V (columns are [3][H][D])
      int c   = n & 1023;
      int h   = c >> 6, d = c & 63;
      float bv = bias[n];
      __bf16* dst = (sec == 0) ? Qo : (sec == 1) ? Ko : Vo;
      float scale = (sec == 0) ? 0.125f : 1.0f;
#pragma unroll
      for (int r = 0; r < 8; ++r) {
        int m  = m0 + wy * 32 + i * 16 + half * 8 + r;
        int bb = m >> 11, t = m & 2047;
        dst[(((size_t)(bb * HH + h)) * TT + t) * DD + d] =
            (__bf16)((acc[i][j][r] + bv) * scale);
      }
    }
}

// ---------------------------------------------------------------------------
// Kernel 2: flash attention (ALiBi + causal).
// 1 WG = (b, h, 64-query block); 4 waves x 16 queries; key blocks of 64.
// K tile fetched with global_load_async_to_lds_b128 (ASYNCcnt path).
// ---------------------------------------------------------------------------
__global__ __launch_bounds__(128) void attn_kernel(
    const __bf16* __restrict__ Q, const __bf16* __restrict__ K,
    const __bf16* __restrict__ V, __bf16* __restrict__ Y)
{
  constexpr int LDK = 72;
  __shared__ __align__(16) __bf16 Ks[64 * LDK];       // [s][d] row-major
  __shared__ __align__(16) __bf16 Vt[64 * LDK];       // transposed [d][s]
  __shared__ __align__(16) __bf16 Ps[4][16 * LDK];    // per-wave P scratch [m][s]

  const int tid  = threadIdx.x;
  const int lane = tid & 31;
  const int wave = tid >> 5;
  const int l16  = lane & 15;
  const int half = lane >> 4;

  const int qb = blockIdx.x;
  const int h  = blockIdx.y;
  const int b  = blockIdx.z;
  const int q0  = qb * 64;
  const int qw0 = q0 + wave * 16;

  const size_t head = ((size_t)(b * HH + h)) * TT;
  const __bf16* Qh = Q + head * DD;
  const __bf16* Kh = K + head * DD;
  const __bf16* Vh = V + head * DD;

  v16bf aq0, aq1;
  {
    const __bf16* qrow = Qh + (size_t)(qw0 + l16) * DD;
    int klo = half * 8;
    ((v8bf*)&aq0)[0] = *(const v8bf*)(qrow + klo);
    ((v8bf*)&aq0)[1] = *(const v8bf*)(qrow + klo + 16);
    ((v8bf*)&aq1)[0] = *(const v8bf*)(qrow + 32 + klo);
    ((v8bf*)&aq1)[1] = *(const v8bf*)(qrow + 32 + klo + 16);
  }

  const float slope = exp2f(-0.5f * (float)(h + 1));   // 2^(-(8/H)*(h+1))

  v8f   o[4] = {};
  float rmax[8], rsum[8];
#pragma unroll
  for (int r = 0; r < 8; ++r) { rmax[r] = -3.0e38f; rsum[r] = 0.0f; }

  const int nkb = qb + 1;
  for (int kb = 0; kb < nkb; ++kb) {
    int s0 = kb * 64;

    // --- K tile: async global->LDS (b128 per lane, 4 chunks per thread) ---
#pragma unroll
    for (int p = 0; p < 4; ++p) {
      int c   = p * 128 + tid;        // 512 chunks of 8 halves
      int s   = c >> 3;
      int off = (c & 7) * 8;
      unsigned ldsa = (unsigned)(uintptr_t)&Ks[s * LDK + off];
      unsigned goff = (unsigned)(((s0 + s) * DD + off) * 2);
      asm volatile("global_load_async_to_lds_b128 %0, %1, %2"
                   :: "v"(ldsa), "v"(goff), "s"(Kh) : "memory");
    }

    // --- V tile: transposed [d][s], packed (s,s+1) pair stores (b32) ---
#pragma unroll
    for (int p = 0; p < 2; ++p) {
      int c   = p * 128 + tid;        // 256 tasks: 32 s-pairs x 8 d-chunks
      int s   = (c >> 3) * 2;
      int off = (c & 7) * 8;
      v8bf va = *(const v8bf*)(Vh + (size_t)(s0 + s)     * DD + off);
      v8bf vb = *(const v8bf*)(Vh + (size_t)(s0 + s + 1) * DD + off);
#pragma unroll
      for (int j = 0; j < 8; ++j)
        st_pk_bf(&Vt[(off + j) * LDK + s], va[j], vb[j]);
    }

    asm volatile("s_wait_asynccnt 0" ::: "memory");
    __syncthreads();

    // S = Q @ K^T
    v8f sfr[4];
#pragma unroll
    for (int j = 0; j < 4; ++j) {
      int n   = j * 16 + l16;
      int kb2 = half * 8;
      v16bf bk0, bk1;
      ((v8bf*)&bk0)[0] = *(const v8bf*)&Ks[n * LDK + kb2];
      ((v8bf*)&bk0)[1] = *(const v8bf*)&Ks[n * LDK + kb2 + 16];
      ((v8bf*)&bk1)[0] = *(const v8bf*)&Ks[n * LDK + 32 + kb2];
      ((v8bf*)&bk1)[1] = *(const v8bf*)&Ks[n * LDK + 32 + kb2 + 16];
      v8f z = {};
      z = WMMA_BF16(aq0, bk0, z);
      z = WMMA_BF16(aq1, bk1, z);
      sfr[j] = z;
    }

    // ALiBi bias (absolute key position) + causal mask in C-fragment layout.
#pragma unroll
    for (int j = 0; j < 4; ++j) {
      int   sg = s0 + j * 16 + l16;
      float bv = slope * (float)sg;
#pragma unroll
      for (int r = 0; r < 8; ++r) {
        int t = qw0 + half * 8 + r;
        float v = sfr[j][r] + bv;
        sfr[j][r] = (sg <= t) ? v : -1.0e30f;
      }
    }

    // Online softmax (row = 16 lanes of one half-wave).
#pragma unroll
    for (int r = 0; r < 8; ++r) {
      float mx = fmaxf(fmaxf(sfr[0][r], sfr[1][r]), fmaxf(sfr[2][r], sfr[3][r]));
#pragma unroll
      for (int off = 1; off < 16; off <<= 1)
        mx = fmaxf(mx, __shfl_xor(mx, off, 32));
      float nm   = fmaxf(rmax[r], mx);
      float corr = __expf(rmax[r] - nm);
      rmax[r] = nm;
      float ls = 0.0f;
#pragma unroll
      for (int j = 0; j < 4; ++j) {
        float pv = __expf(sfr[j][r] - nm);
        sfr[j][r] = pv;
        ls += pv;
      }
#pragma unroll
      for (int off = 1; off < 16; off <<= 1)
        ls += __shfl_xor(ls, off, 32);
      rsum[r] = rsum[r] * corr + ls;
#pragma unroll
      for (int jd = 0; jd < 4; ++jd) o[jd][r] *= corr;
    }

    // Transpose P (C-layout -> A-layout) through per-wave LDS scratch.
#pragma unroll
    for (int j = 0; j < 4; ++j)
#pragma unroll
      for (int r = 0; r < 8; ++r)
        Ps[wave][(half * 8 + r) * LDK + j * 16 + l16] = (__bf16)sfr[j][r];

    v16bf ap0, ap1;
    {
      const __bf16* pr = &Ps[wave][l16 * LDK];
      int klo = half * 8;
      ((v8bf*)&ap0)[0] = *(const v8bf*)(pr + klo);
      ((v8bf*)&ap0)[1] = *(const v8bf*)(pr + klo + 16);
      ((v8bf*)&ap1)[0] = *(const v8bf*)(pr + 32 + klo);
      ((v8bf*)&ap1)[1] = *(const v8bf*)(pr + 32 + klo + 16);
    }

    // O += P @ V
#pragma unroll
    for (int jd = 0; jd < 4; ++jd) {
      int d   = jd * 16 + l16;
      int kb2 = half * 8;
      v16bf bv0, bv1;
      ((v8bf*)&bv0)[0] = *(const v8bf*)&Vt[d * LDK + kb2];
      ((v8bf*)&bv0)[1] = *(const v8bf*)&Vt[d * LDK + kb2 + 16];
      ((v8bf*)&bv1)[0] = *(const v8bf*)&Vt[d * LDK + 32 + kb2];
      ((v8bf*)&bv1)[1] = *(const v8bf*)&Vt[d * LDK + 32 + kb2 + 16];
      o[jd] = WMMA_BF16(ap0, bv0, o[jd]);
      o[jd] = WMMA_BF16(ap1, bv1, o[jd]);
    }
    __syncthreads();
  }

  // Normalize and store Y[b][t][h*64+d] bf16.
#pragma unroll
  for (int r = 0; r < 8; ++r) {
    float inv = 1.0f / rsum[r];
    int t = qw0 + half * 8 + r;
    __bf16* yrow = Y + ((size_t)(b * TT + t)) * CC_ + h * DD;
#pragma unroll
    for (int jd = 0; jd < 4; ++jd)
      yrow[jd * 16 + l16] = (__bf16)(o[jd][r] * inv);
  }
}

// ---------------------------------------------------------------------------
// Kernel 3: out = Y @ w_proj + b_proj (f32 out). 128x128 block, 32x64 waves.
// ---------------------------------------------------------------------------
__global__ __launch_bounds__(256) void proj_gemm_kernel(
    const __bf16* __restrict__ Y, const float* __restrict__ w,
    const float* __restrict__ bias, float* __restrict__ out)
{
  constexpr int LDA = 40, LDB = 40;
  __shared__ __align__(16) __bf16 As[128 * LDA];
  __shared__ __align__(16) __bf16 Bs[128 * LDB];

  const int tid  = threadIdx.x;
  const int lane = tid & 31;
  const int wave = tid >> 5;
  const int wy   = wave & 3;
  const int wx   = wave >> 2;
  const int l16  = lane & 15;
  const int half = lane >> 4;
  const int m0 = blockIdx.y * 128;
  const int n0 = blockIdx.x * 128;

  v8f acc[2][4] = {};

  for (int kk = 0; kk < 1024; kk += 32) {
    // A: 128x32 bf16, direct b128 copies.
#pragma unroll
    for (int p = 0; p < 2; ++p) {
      int c   = p * 256 + tid;
      int row = c >> 2;
      int off = (c & 3) * 8;
      *(v8bf*)&As[row * LDA + off] =
          *(const v8bf*)(Y + (size_t)(m0 + row) * 1024 + kk + off);
    }
    // B: 32x128 f32, transposed [n][k] with packed pair stores.
#pragma unroll
    for (int p = 0; p < 2; ++p) {
      int c   = p * 256 + tid;
      int kp  = c >> 5;
      int col = (c & 31) * 4;
      const float* wp = w + (size_t)(kk + 2 * kp) * 1024 + (n0 + col);
      v4f r0 = *(const v4f*)wp;
      v4f r1 = *(const v4f*)(wp + 1024);
#pragma unroll
      for (int j = 0; j < 4; ++j)
        st_pk_f(&Bs[(col + j) * LDB + 2 * kp], r0[j], r1[j]);
    }
    __syncthreads();

    v16bf afr[2], bfr[4];
#pragma unroll
    for (int i = 0; i < 2; ++i) {
      int m   = wy * 32 + i * 16 + l16;
      int klo = half * 8;
      ((v8bf*)&afr[i])[0] = *(const v8bf*)&As[m * LDA + klo];
      ((v8bf*)&afr[i])[1] = *(const v8bf*)&As[m * LDA + klo + 16];
    }
#pragma unroll
    for (int j = 0; j < 4; ++j) {
      int n  = wx * 64 + j * 16 + l16;
      int kb = half * 8;
      ((v8bf*)&bfr[j])[0] = *(const v8bf*)&Bs[n * LDB + kb];
      ((v8bf*)&bfr[j])[1] = *(const v8bf*)&Bs[n * LDB + kb + 16];
    }
#pragma unroll
    for (int i = 0; i < 2; ++i)
#pragma unroll
      for (int j = 0; j < 4; ++j)
        acc[i][j] = WMMA_BF16(afr[i], bfr[j], acc[i][j]);

    __syncthreads();
  }

#pragma unroll
  for (int i = 0; i < 2; ++i)
#pragma unroll
    for (int j = 0; j < 4; ++j) {
      int n = n0 + wx * 64 + j * 16 + l16;
      float bv = bias[n];
#pragma unroll
      for (int r = 0; r < 8; ++r) {
        int m = m0 + wy * 32 + i * 16 + half * 8 + r;
        out[(size_t)m * 1024 + n] = acc[i][j][r] + bv;
      }
    }
}

// ---------------------------------------------------------------------------
extern "C" void kernel_launch(void* const* d_in, const int* in_sizes, int n_in,
                              void* d_out, int out_size, void* d_ws, size_t ws_size,
                              hipStream_t stream) {
  const float* x      = (const float*)d_in[0];
  const float* w_attn = (const float*)d_in[1];
  const float* b_attn = (const float*)d_in[2];
  const float* w_proj = (const float*)d_in[3];
  const float* b_proj = (const float*)d_in[4];
  float* out = (float*)d_out;

  const size_t HEADSZ = (size_t)2 * HH * TT * DD;   // 4,194,304 halves
  __bf16* Q = (__bf16*)d_ws;
  __bf16* K = Q + HEADSZ;
  __bf16* V = K + HEADSZ;
  __bf16* Y = V + HEADSZ;

  qkv_gemm_kernel<<<dim3(24, 32), 256, 0, stream>>>(x, w_attn, b_attn, Q, K, V);
  attn_kernel<<<dim3(TT / 64, HH, 2), 128, 0, stream>>>(Q, K, V, Y);
  proj_gemm_kernel<<<dim3(8, 32), 256, 0, stream>>>(Y, w_proj, b_proj, out);
}
